// VLMDistillGVendi_17317308137684
// MI455X (gfx1250) — matrix operand order repaired and checked by reading, not compiled, not run on hardware
//
#include <hip/hip_runtime.h>
#include <math.h>

// ---------------- constants ----------------
#define B_N     65536
#define D_DIM   256
#define KF      60
#define KF_PAD  64
#define KC      40
#define KC_PAD  48
#define REG_F   0.05f
#define INV_REG 20.0f
#define N_ITERS 100
#define LOG_A   (-11.090354888959125f)   // -log(65536)
#define LOG_B_F (-4.0943445622221004f)   // -log(60)
#define LOG_B_C (-3.6888794541139363f)   // -log(40)
#define LDS_STRIDE 264                    // ushorts per staged row (bank-conflict padding)

typedef __attribute__((ext_vector_type(16))) __bf16 v16bf;
typedef __attribute__((ext_vector_type(8)))  float  v8f;

union BF16Frag { v16bf v; __bf16 h[16]; unsigned short s[16]; uint4 q[2]; };

// hardware f32->bf16 (backend selects v_cvt_pk_bf16_f32 on gfx1250)
__device__ __forceinline__ unsigned short cvt_bf16(float x) {
  union { __bf16 b; unsigned short u; } r;
  r.b = (__bf16)x;
  return r.u;
}

__device__ __forceinline__ v8f wmma_bf16(v16bf a, v16bf b, v8f c) {
  return __builtin_amdgcn_wmma_f32_16x16x32_bf16(false, a, false, b, (short)0, c, false, false);
}

__device__ __forceinline__ v8f zero_v8f() {
  v8f a;
#pragma unroll
  for (int i = 0; i < 8; ++i) a[i] = 0.0f;
  return a;
}

// A fragment (16x32 bf16) from bf16 row-major [rows][256] in global memory.
// Layout: lane<16 holds row M=lane, K = {0..15}; lane>=16: K = {16..31} (matches B fragments)
__device__ __forceinline__ v16bf load_afrag_bf16(const unsigned short* X, int r0, int k0) {
  int lane = threadIdx.x & 31;
  int hl = lane & 15, hi = lane >> 4;
  const unsigned short* p = X + (size_t)(r0 + hl) * D_DIM + k0 + hi * 16;
  BF16Frag f;
  f.q[0] = *(const uint4*)(p);
  f.q[1] = *(const uint4*)(p + 8);
  return f.v;
}

// A fragment from LDS-staged bf16 tile (row stride LDS_STRIDE ushorts)
__device__ __forceinline__ v16bf load_afrag_lds(const unsigned short* sA, int k0) {
  int lane = threadIdx.x & 31;
  int hl = lane & 15, hi = lane >> 4;
  const unsigned short* p = sA + hl * LDS_STRIDE + k0 + hi * 16;
  BF16Frag f;
  f.q[0] = *(const uint4*)(p);
  f.q[1] = *(const uint4*)(p + 8);
  return f.v;
}

// B fragment (32x16 bf16) from pre-converted bf16 matrix: B[k][n] = X[n0+hl][k0+hi*16+0..15]
__device__ __forceinline__ v16bf load_bfrag_bf16(const unsigned short* X, int n0, int k0) {
  int lane = threadIdx.x & 31;
  int hl = lane & 15, hi = lane >> 4;
  const unsigned short* p = X + (size_t)(n0 + hl) * D_DIM + k0 + hi * 16;
  BF16Frag f;
  f.q[0] = *(const uint4*)(p);
  f.q[1] = *(const uint4*)(p + 8);
  return f.v;
}

// B fragment from f32 row-major, scaled then converted (contiguous 16 elems per lane)
__device__ __forceinline__ v16bf load_bfrag_f32(const float* X, int n0, int k0, float sc) {
  int lane = threadIdx.x & 31;
  int hl = lane & 15, hi = lane >> 4;
  const float* p = X + (size_t)(n0 + hl) * D_DIM + k0 + hi * 16;
  float4 a0 = *(const float4*)(p);
  float4 a1 = *(const float4*)(p + 4);
  float4 a2 = *(const float4*)(p + 8);
  float4 a3 = *(const float4*)(p + 12);
  BF16Frag f;
  f.s[0]  = cvt_bf16(a0.x*sc); f.s[1]  = cvt_bf16(a0.y*sc); f.s[2]  = cvt_bf16(a0.z*sc); f.s[3]  = cvt_bf16(a0.w*sc);
  f.s[4]  = cvt_bf16(a1.x*sc); f.s[5]  = cvt_bf16(a1.y*sc); f.s[6]  = cvt_bf16(a1.z*sc); f.s[7]  = cvt_bf16(a1.w*sc);
  f.s[8]  = cvt_bf16(a2.x*sc); f.s[9]  = cvt_bf16(a2.y*sc); f.s[10] = cvt_bf16(a2.z*sc); f.s[11] = cvt_bf16(a2.w*sc);
  f.s[12] = cvt_bf16(a3.x*sc); f.s[13] = cvt_bf16(a3.y*sc); f.s[14] = cvt_bf16(a3.z*sc); f.s[15] = cvt_bf16(a3.w*sc);
  return f.v;
}

__device__ __forceinline__ float wave_sum32(float v) {
#pragma unroll
  for (int m = 16; m > 0; m >>= 1) v += __shfl_xor(v, m, 32);
  return v;
}

// ---------------- kernel 0: init tiny state ----------------
__global__ void init_kernel(float* logv_f, float* logv_c) {
  int t = threadIdx.x;
  if (t < KF_PAD) logv_f[t] = 0.0f;
  if (t < KC_PAD) logv_c[t] = 0.0f;
}

// ---------------- kernel 1: per-row 1/||x|| for the six BxD matrices ----------------
__global__ __launch_bounds__(256) void rownorm_kernel(
    const float* g0, const float* g1, const float* g2,
    const float* g3, const float* g4, const float* g5, float* rnorm) {
  int wave = threadIdx.x >> 5, lane = threadIdx.x & 31;
  int mat = blockIdx.y;
  int row = blockIdx.x * 8 + wave;
  const float* X = (mat == 0) ? g0 : (mat == 1) ? g1 : (mat == 2) ? g2
                 : (mat == 3) ? g3 : (mat == 4) ? g4 : g5;
  const float* p = X + (size_t)row * D_DIM + lane * 8;
  float4 a = *(const float4*)(p);
  float4 b = *(const float4*)(p + 4);
  float s = a.x*a.x + a.y*a.y + a.z*a.z + a.w*a.w
          + b.x*b.x + b.y*b.y + b.z*b.z + b.w*b.w;
  s = wave_sum32(s);
  if (lane == 0)
    rnorm[(size_t)mat * B_N + row] = 1.0f / fmaxf(sqrtf(s), 1e-12f);
}

// ---------------- kernel 2: normalize codebooks, zero-pad, emit f32 + bf16 ----------------
__global__ __launch_bounds__(256) void centnorm_kernel(
    const float* book, const float* vcent, const float* tcent,
    float* Cf32, float* Vf32, float* Tf32,
    unsigned short* Cbf, unsigned short* Vbf, unsigned short* Tbf) {
  int wave = threadIdx.x >> 5, lane = threadIdx.x & 31;
  int slot = blockIdx.x * 8 + wave;            // 0..159
  const float* src; float* dF; unsigned short* dB; int row; int kreal;
  if (slot < KF_PAD)              { src = book;  dF = Cf32; dB = Cbf; row = slot;            kreal = KF; }
  else if (slot < KF_PAD + KC_PAD){ src = vcent; dF = Vf32; dB = Vbf; row = slot - KF_PAD;   kreal = KC; }
  else                            { src = tcent; dF = Tf32; dB = Tbf; row = slot - KF_PAD - KC_PAD; kreal = KC; }
  float* of = dF + (size_t)row * D_DIM + lane * 8;
  unsigned short* ob = dB + (size_t)row * D_DIM + lane * 8;
  if (row < kreal) {
    const float* p = src + (size_t)row * D_DIM + lane * 8;
    float4 a = *(const float4*)(p);
    float4 b = *(const float4*)(p + 4);
    float s = a.x*a.x + a.y*a.y + a.z*a.z + a.w*a.w
            + b.x*b.x + b.y*b.y + b.z*b.z + b.w*b.w;
    s = wave_sum32(s);
    float sc = 1.0f / fmaxf(sqrtf(s), 1e-12f);
    float v[8] = {a.x*sc, a.y*sc, a.z*sc, a.w*sc, b.x*sc, b.y*sc, b.z*sc, b.w*sc};
#pragma unroll
    for (int t = 0; t < 8; ++t) { of[t] = v[t]; ob[t] = cvt_bf16(v[t]); }
  } else {
#pragma unroll
    for (int t = 0; t < 8; ++t) { of[t] = 0.0f; ob[t] = 0; }
  }
}

// ---------------- kernel 3: convert a DxD weight matrix to bf16 ----------------
__global__ __launch_bounds__(256) void convW_kernel(
    const float* W0, const float* W1, const float* W2, unsigned short* Wbf) {
  int wave = threadIdx.x >> 5, lane = threadIdx.x & 31;
  int mat = blockIdx.y;
  const float* W = (mat == 0) ? W0 : (mat == 1) ? W1 : W2;
  int row = blockIdx.x * 8 + wave;
  const float* p = W + (size_t)row * D_DIM + lane * 8;
  unsigned short* o = Wbf + (size_t)mat * D_DIM * D_DIM + (size_t)row * D_DIM + lane * 8;
  float4 a = *(const float4*)(p);
  float4 b = *(const float4*)(p + 4);
  o[0] = cvt_bf16(a.x); o[1] = cvt_bf16(a.y); o[2] = cvt_bf16(a.z); o[3] = cvt_bf16(a.w);
  o[4] = cvt_bf16(b.x); o[5] = cvt_bf16(b.y); o[6] = cvt_bf16(b.z); o[7] = cvt_bf16(b.w);
}

// ---------------- kernel 4: fusion cost GEMM -> MTf[k][i] (transposed, /REG) ----------------
// All 4 m-tiles kept in registers: gt_f fragment converted once, feeds 4 WMMAs.
__global__ __launch_bounds__(256) void cost_fusion_kernel(
    const float* gt_f, const float* rnorm, const unsigned short* Cbf, float* MT) {
  int wave = threadIdx.x >> 5, lane = threadIdx.x & 31;
  int hl = lane & 15, hi = lane >> 4;
  int i0 = (blockIdx.x * 8 + wave) * 16;       // batch rows
  float sc = rnorm[2 * B_N + i0 + hl];         // gt_f is matrix index 2
  v8f acc[4];
#pragma unroll
  for (int mt = 0; mt < 4; ++mt) acc[mt] = zero_v8f();
#pragma unroll
  for (int kk = 0; kk < 8; ++kk) {
    int k0 = kk * 32;
    v16bf b = load_bfrag_f32(gt_f, i0, k0, sc);
#pragma unroll
    for (int mt = 0; mt < 4; ++mt) {
      v16bf a = load_afrag_bf16(Cbf, mt * 16, k0);
      acc[mt] = wmma_bf16(a, b, acc[mt]);
    }
  }
  int i = i0 + hl;
#pragma unroll
  for (int mt = 0; mt < 4; ++mt) {
#pragma unroll
    for (int v = 0; v < 8; ++v) {
      int m = mt * 16 + v + hi * 8;
      if (m < KF) {
        float cost = fmaxf(2.0f - 2.0f * acc[mt][v], 0.0f);
        MT[(size_t)m * B_N + i] = cost * INV_REG;
      }
    }
  }
}

// ---------------- kernel 5: cross cost GEMM -> MTc[k][i] ----------------
__global__ __launch_bounds__(256) void cost_cross_kernel(
    const float* gt_v, const float* gt_t, const float* rnorm,
    const unsigned short* Vbf, const unsigned short* Tbf, float* MT) {
  int wave = threadIdx.x >> 5, lane = threadIdx.x & 31;
  int hl = lane & 15, hi = lane >> 4;
  int i0 = (blockIdx.x * 8 + wave) * 16;
  float sv = rnorm[0 * B_N + i0 + hl];
  float st = rnorm[1 * B_N + i0 + hl];
  v8f accv[3], acct[3];
#pragma unroll
  for (int mt = 0; mt < 3; ++mt) { accv[mt] = zero_v8f(); acct[mt] = zero_v8f(); }
#pragma unroll
  for (int kk = 0; kk < 8; ++kk) {
    int k0 = kk * 32;
    v16bf bv = load_bfrag_f32(gt_v, i0, k0, sv);
    v16bf bt = load_bfrag_f32(gt_t, i0, k0, st);
#pragma unroll
    for (int mt = 0; mt < 3; ++mt) {
      v16bf av = load_afrag_bf16(Vbf, mt * 16, k0);
      accv[mt] = wmma_bf16(av, bv, accv[mt]);
      v16bf at = load_afrag_bf16(Tbf, mt * 16, k0);
      acct[mt] = wmma_bf16(at, bt, acct[mt]);
    }
  }
  int i = i0 + hl;
#pragma unroll
  for (int mt = 0; mt < 3; ++mt) {
#pragma unroll
    for (int v = 0; v < 8; ++v) {
      int m = mt * 16 + v + hi * 8;
      if (m < KC) {
        float cv = fmaxf(2.0f - 2.0f * accv[mt][v], 0.0f);
        float ct = fmaxf(2.0f - 2.0f * acct[mt][v], 0.0f);
        MT[(size_t)m * B_N + i] = (0.5f * cv + 0.5f * ct) * INV_REG;
      }
    }
  }
}

// ---------------- kernel 6: per-row argmin over transposed cost ----------------
__global__ __launch_bounds__(256) void argmin_kernel(const float* MT, int K, int* ksel) {
  int i = blockIdx.x * 256 + threadIdx.x;
  float best = MT[i]; int bk = 0;
  for (int j = 1; j < K; ++j) {
    float v = MT[(size_t)j * B_N + i];
    if (v < best) { best = v; bk = j; }
  }
  ksel[i] = bk;
}

// ---------------- kernel 7: Sinkhorn row pass (log_u update), both problems ----------------
__global__ __launch_bounds__(256) void sink_row_kernel(
    const float* MTf, const float* MTc, const float* logv_f, const float* logv_c,
    float* logu_f, float* logu_c) {
  __shared__ float lv[KF_PAD];
  int prob = blockIdx.y;
  const float* MT   = prob ? MTc    : MTf;
  const float* logv = prob ? logv_c : logv_f;
  float*       logu = prob ? logu_c : logu_f;
  int K = prob ? KC : KF;
  if (threadIdx.x < K) lv[threadIdx.x] = logv[threadIdx.x];
  __syncthreads();
  int i = blockIdx.x * 256 + threadIdx.x;
  float mx = -3.0e38f, s = 0.0f;
  for (int j = 0; j < K; ++j) {
    float t = lv[j] - MT[(size_t)j * B_N + i];
    if (t > mx) { s = s * __expf(mx - t) + 1.0f; mx = t; }
    else        { s += __expf(t - mx); }
  }
  logu[i] = LOG_A - (mx + __logf(s));
}

// ---------------- kernel 8: Sinkhorn column pass (log_v update), block per column ----------------
__global__ __launch_bounds__(256) void sink_col_kernel(
    const float* MTf, const float* MTc, const float* logu_f, const float* logu_c,
    float* logv_f, float* logv_c) {
  __shared__ float smx[256], ssm[256];
  int j = blockIdx.x;
  const float* MT; const float* logu; float* logv; float logb; int col;
  if (j < KF) { MT = MTf; logu = logu_f; logv = logv_f; col = j;      logb = LOG_B_F; }
  else        { MT = MTc; logu = logu_c; logv = logv_c; col = j - KF; logb = LOG_B_C; }
  const float* Mrow = MT + (size_t)col * B_N;
  float mx = -3.0e38f, s = 0.0f;
  for (int i = threadIdx.x; i < B_N; i += 256) {
    float t = logu[i] - Mrow[i];
    if (t > mx) { s = s * __expf(mx - t) + 1.0f; mx = t; }
    else        { s += __expf(t - mx); }
  }
  smx[threadIdx.x] = mx; ssm[threadIdx.x] = s;
  __syncthreads();
  for (int off = 128; off > 0; off >>= 1) {
    if (threadIdx.x < off) {
      float m1 = smx[threadIdx.x], s1 = ssm[threadIdx.x];
      float m2 = smx[threadIdx.x + off], s2 = ssm[threadIdx.x + off];
      if (m2 > m1) { s1 = s1 * __expf(m1 - m2) + s2; m1 = m2; }
      else         { s1 += s2 * __expf(m2 - m1); }
      smx[threadIdx.x] = m1; ssm[threadIdx.x] = s1;
    }
    __syncthreads();
  }
  if (threadIdx.x == 0) logv[col] = logb - (smx[0] + __logf(ssm[0]));
}

// ---------------- kernel 9: L_ot = sum(gamma * cost), block per column -> partial ----------------
__global__ __launch_bounds__(256) void lot_kernel(
    const float* MTf, const float* MTc, const float* logu_f, const float* logu_c,
    const float* logv_f, const float* logv_c, float* lot_part) {
  __shared__ float red[256];
  int j = blockIdx.x;
  const float* MT; const float* logu; const float* logv; int col;
  if (j < KF) { MT = MTf; logu = logu_f; logv = logv_f; col = j; }
  else        { MT = MTc; logu = logu_c; logv = logv_c; col = j - KF; }
  const float* Mrow = MT + (size_t)col * B_N;
  float lv = logv[col];
  float acc = 0.0f;
  for (int i = threadIdx.x; i < B_N; i += 256) {
    float M = Mrow[i];
    acc += __expf(logu[i] + lv - M) * (M * REG_F);
  }
  red[threadIdx.x] = acc;
  __syncthreads();
  for (int off = 128; off > 0; off >>= 1) {
    if (threadIdx.x < off) red[threadIdx.x] += red[threadIdx.x + off];
    __syncthreads();
  }
  if (threadIdx.x == 0) lot_part[j] = red[0];
}

// ---------------- kernel 10: projection GEMM + fused commitment epilogue ----------------
// P = normalize(gs) @ W^T. Block = 16 batch rows x 256 output cols. The normalized-bf16
// 16x256 gs tile is staged once through LDS; W is pre-converted bf16 (zero-VALU fragments).
__global__ __launch_bounds__(256) void proj_commit_kernel(
    const float* gs, const float* rn, const unsigned short* Wbf,
    const float* Cn, const int* ksel, float* part) {
  __shared__ unsigned short sA[16 * LDS_STRIDE];
  __shared__ float sPP[8 * 16], sPC[8 * 16], sCC[8 * 16];
  __shared__ float sROW[16];
  int i0 = blockIdx.x * 16;

  // ---- phase 1: stage normalized bf16 tile (each thread converts 16 elements once) ----
  {
    int r  = threadIdx.x >> 4;          // 0..15
    int c0 = (threadIdx.x & 15) * 16;   // 0..240
    float sc = rn[i0 + r];
    const float* p = gs + (size_t)(i0 + r) * D_DIM + c0;
    float4 x0 = *(const float4*)(p);
    float4 x1 = *(const float4*)(p + 4);
    float4 x2 = *(const float4*)(p + 8);
    float4 x3 = *(const float4*)(p + 12);
    BF16Frag f;
    f.s[0]  = cvt_bf16(x0.x*sc); f.s[1]  = cvt_bf16(x0.y*sc); f.s[2]  = cvt_bf16(x0.z*sc); f.s[3]  = cvt_bf16(x0.w*sc);
    f.s[4]  = cvt_bf16(x1.x*sc); f.s[5]  = cvt_bf16(x1.y*sc); f.s[6]  = cvt_bf16(x1.z*sc); f.s[7]  = cvt_bf16(x1.w*sc);
    f.s[8]  = cvt_bf16(x2.x*sc); f.s[9]  = cvt_bf16(x2.y*sc); f.s[10] = cvt_bf16(x2.z*sc); f.s[11] = cvt_bf16(x2.w*sc);
    f.s[12] = cvt_bf16(x3.x*sc); f.s[13] = cvt_bf16(x3.y*sc); f.s[14] = cvt_bf16(x3.z*sc); f.s[15] = cvt_bf16(x3.w*sc);
    uint4* d = (uint4*)(sA + r * LDS_STRIDE + c0);
    d[0] = f.q[0];
    d[1] = f.q[1];
  }
  __syncthreads();

  // ---- phase 2: WMMA GEMM, 2 N-tiles per wave ----
  int wave = threadIdx.x >> 5, lane = threadIdx.x & 31;
  int hl = lane & 15, hi = lane >> 4;
  int n0 = wave * 32;
  v8f acc0 = zero_v8f();
  v8f acc1 = zero_v8f();
#pragma unroll
  for (int kk = 0; kk < 8; ++kk) {
    int k0 = kk * 32;
    v16bf a  = load_afrag_lds(sA, k0);
    v16bf b0 = load_bfrag_bf16(Wbf, n0,      k0);
    v16bf b1 = load_bfrag_bf16(Wbf, n0 + 16, k0);
    acc0 = wmma_bf16(a, b0, acc0);
    acc1 = wmma_bf16(a, b1, acc1);
  }

  // ---- phase 3: fused commitment epilogue ----
#pragma unroll
  for (int v = 0; v < 8; ++v) {
    int m = v + hi * 8;
    int kidx = ksel[i0 + m];
    const float* crow = Cn + (size_t)kidx * D_DIM;
    float p0 = acc0[v], p1 = acc1[v];
    float c0 = crow[n0 + hl], c1 = crow[n0 + 16 + hl];
    float pp = p0 * p0 + p1 * p1;
    float pc = p0 * c0 + p1 * c1;
    float cc = c0 * c0 + c1 * c1;
#pragma unroll
    for (int msk = 8; msk; msk >>= 1) {      // reduce across the 16 lanes of this half-wave
      pp += __shfl_xor(pp, msk, 32);
      pc += __shfl_xor(pc, msk, 32);
      cc += __shfl_xor(cc, msk, 32);
    }
    if (hl == 0) {
      sPP[wave * 16 + m] = pp;
      sPC[wave * 16 + m] = pc;
      sCC[wave * 16 + m] = cc;
    }
  }
  __syncthreads();
  if (threadIdx.x < 16) {
    int m = threadIdx.x;
    float pp = 0.0f, pc = 0.0f, cc = 0.0f;
    for (int w = 0; w < 8; ++w) { pp += sPP[w * 16 + m]; pc += sPC[w * 16 + m]; cc += sCC[w * 16 + m]; }
    sROW[m] = pp - 2.0f * pc + cc;
  }
  __syncthreads();
  if (threadIdx.x == 0) {
    float s = 0.0f;
    for (int m = 0; m < 16; ++m) s += sROW[m];
    part[blockIdx.x] = s;
  }
}

// ---------------- kernel 11: deterministic final reduction ----------------
__global__ __launch_bounds__(256) void finalize_kernel(
    const float* lot_part, const float* commit_part, float* out) {
  __shared__ float red[256];
  int t = threadIdx.x;
  float lf = 0.0f, lc = 0.0f, cf = 0.0f, cv = 0.0f, ct = 0.0f;
  for (int j = t; j < KF; j += 256) lf += lot_part[j];
  for (int j = KF + t; j < KF + KC; j += 256) lc += lot_part[j];
  for (int j = t; j < 4096; j += 256) {
    cf += commit_part[j];
    cv += commit_part[4096 + j];
    ct += commit_part[8192 + j];
  }
  float tot = 1.2f * (lf + cf) + 0.5f * (lc + 0.5f * cv + 0.5f * ct);
  red[t] = tot;
  __syncthreads();
  for (int off = 128; off > 0; off >>= 1) {
    if (t < off) red[t] += red[t + off];
    __syncthreads();
  }
  if (t == 0) out[0] = red[0];
}

// ---------------- host launcher ----------------
extern "C" void kernel_launch(void* const* d_in, const int* in_sizes, int n_in,
                              void* d_out, int out_size, void* d_ws, size_t ws_size,
                              hipStream_t stream) {
  const float* gt_v  = (const float*)d_in[0];
  const float* gt_t  = (const float*)d_in[1];
  const float* gt_f  = (const float*)d_in[2];
  const float* gs_v  = (const float*)d_in[3];
  const float* gs_t  = (const float*)d_in[4];
  const float* gs_f  = (const float*)d_in[5];
  const float* W_v   = (const float*)d_in[6];
  const float* W_t   = (const float*)d_in[7];
  const float* W_f   = (const float*)d_in[8];
  const float* book  = (const float*)d_in[9];
  const float* vcent = (const float*)d_in[10];
  const float* tcent = (const float*)d_in[11];
  float* out = (float*)d_out;

  char* ws = (char*)d_ws;
  size_t off = 0;
  auto carve = [&](size_t bytes) -> void* {
    void* p = ws + off;
    off = (off + bytes + 255) & ~(size_t)255;
    return p;
  };
  float* rnorm   = (float*)carve((size_t)6 * B_N * 4);
  float* MTf     = (float*)carve((size_t)KF_PAD * B_N * 4);
  float* MTc     = (float*)carve((size_t)KC_PAD * B_N * 4);
  float* logu_f  = (float*)carve((size_t)B_N * 4);
  float* logu_c  = (float*)carve((size_t)B_N * 4);
  float* logv_f  = (float*)carve(KF_PAD * 4);
  float* logv_c  = (float*)carve(KC_PAD * 4);
  int*   kf      = (int*)carve((size_t)B_N * 4);
  int*   kc      = (int*)carve((size_t)B_N * 4);
  float* Cf32    = (float*)carve((size_t)KF_PAD * D_DIM * 4);
  float* Vf32    = (float*)carve((size_t)KC_PAD * D_DIM * 4);
  float* Tf32    = (float*)carve((size_t)KC_PAD * D_DIM * 4);
  unsigned short* Cbf = (unsigned short*)carve((size_t)KF_PAD * D_DIM * 2);
  unsigned short* Vbf = (unsigned short*)carve((size_t)KC_PAD * D_DIM * 2);
  unsigned short* Tbf = (unsigned short*)carve((size_t)KC_PAD * D_DIM * 2);
  unsigned short* Wbf = (unsigned short*)carve((size_t)3 * D_DIM * D_DIM * 2);
  float* lot_part    = (float*)carve(128 * 4);
  float* commit_part = (float*)carve((size_t)3 * 4096 * 4);
  if (off > ws_size) return;  // workspace too small; refuse to scribble

  init_kernel<<<1, 256, 0, stream>>>(logv_f, logv_c);
  rownorm_kernel<<<dim3(B_N / 8, 6), 256, 0, stream>>>(gt_v, gt_t, gt_f, gs_v, gs_t, gs_f, rnorm);
  centnorm_kernel<<<20, 256, 0, stream>>>(book, vcent, tcent, Cf32, Vf32, Tf32, Cbf, Vbf, Tbf);
  convW_kernel<<<dim3(D_DIM / 8, 3), 256, 0, stream>>>(W_v, W_t, W_f, Wbf);

  cost_fusion_kernel<<<B_N / 128, 256, 0, stream>>>(gt_f, rnorm, Cbf, MTf);
  cost_cross_kernel<<<B_N / 128, 256, 0, stream>>>(gt_v, gt_t, rnorm, Vbf, Tbf, MTc);

  argmin_kernel<<<B_N / 256, 256, 0, stream>>>(MTf, KF, kf);
  argmin_kernel<<<B_N / 256, 256, 0, stream>>>(MTc, KC, kc);

  for (int it = 0; it < N_ITERS; ++it) {
    sink_row_kernel<<<dim3(B_N / 256, 2), 256, 0, stream>>>(MTf, MTc, logv_f, logv_c, logu_f, logu_c);
    sink_col_kernel<<<KF + KC, 256, 0, stream>>>(MTf, MTc, logu_f, logu_c, logv_f, logv_c);
  }

  lot_kernel<<<KF + KC, 256, 0, stream>>>(MTf, MTc, logu_f, logu_c, logv_f, logv_c, lot_part);

  proj_commit_kernel<<<B_N / 16, 256, 0, stream>>>(gs_f, rnorm + (size_t)5 * B_N, Wbf,                       Cf32, kf, commit_part);
  proj_commit_kernel<<<B_N / 16, 256, 0, stream>>>(gs_v, rnorm + (size_t)3 * B_N, Wbf + (size_t)D_DIM * D_DIM,     Vf32, kc, commit_part + 4096);
  proj_commit_kernel<<<B_N / 16, 256, 0, stream>>>(gs_t, rnorm + (size_t)4 * B_N, Wbf + (size_t)2 * D_DIM * D_DIM, Tf32, kc, commit_part + 8192);

  finalize_kernel<<<1, 256, 0, stream>>>(lot_part, commit_part, out);
}
